// LR_PINN_phase2_midout_87497073754924
// MI455X (gfx1250) — matrix operand order repaired and checked by reading, not compiled, not run on hardware
//
#include <hip/hip_runtime.h>
#include <math.h>

// ---- CDNA5 WMMA types -------------------------------------------------------
typedef __bf16 bf16;
typedef __attribute__((ext_vector_type(16))) __bf16        v16bf;
typedef __attribute__((ext_vector_type(8)))  float         v8f;
typedef __attribute__((ext_vector_type(4)))  unsigned int  u32x4;

#define WAVES    4
#define HPITCH   264   // bf16 elems per h-tile row: 256 + 8 pad (row stride 528B, 16B aligned, bank-spread)
#define WCPITCH  264   // WcT (= (col*alpha)^T, [32][256]) row pitch in bf16
#define WRPITCH  40    // Wr ([256][32]) row pitch in bf16 (80B, 16B aligned)
#define UPITCH   40    // U staging tile pitch

// LDS carve-out (bytes) -- dual 16-row tiles per wave (M=32 per wave)
#define SZ_WC   (3 * 32 * WCPITCH * 2)            // 50688
#define SZ_WR   (3 * 256 * WRPITCH * 2)           // 61440
#define SZ_H    (WAVES * 32 * HPITCH * 2)         // 67584
#define SZ_U    (WAVES * 32 * UPITCH * 2)         // 10240
#define OFF_WC  0
#define OFF_WR  (OFF_WC + SZ_WC)
#define OFF_H   (OFF_WR + SZ_WR)
#define OFF_U   (OFF_H + SZ_H)
#define SMEM_BYTES (OFF_U + SZ_U)                 // 189952 B (< 320KB/WG on CDNA5)

union V16U { u32x4 q[2]; v16bf v; };
union PK2  { bf16 h[2]; unsigned int u; };

// Hardware tanh: single TRANS op on gfx1250 (co-executes with XDL WMMA).
__device__ __forceinline__ float fast_tanh(float v) {
#if __has_builtin(__builtin_amdgcn_tanhf)
  return __builtin_amdgcn_tanhf(v);
#else
  float r;
  asm volatile("v_tanh_f32 %0, %1\n\tv_nop" : "=v"(r) : "v"(v));
  return r;
#endif
}

// Build a 16-element bf16 fragment from two 16B LDS chunks (two ds_load_b128).
__device__ __forceinline__ v16bf ld_frag(const char* p0, const char* p1) {
  V16U u;
  u.q[0] = *(const u32x4*)p0;
  u.q[1] = *(const u32x4*)p1;
  return u.v;
}

__device__ __forceinline__ v8f wmma_bf16(v16bf a, v16bf b, v8f c) {
  // D(f32 16x16) = A(bf16 16x32) * B(bf16 32x16) + C
  return __builtin_amdgcn_wmma_f32_16x16x32_bf16(false, a, false, b, (short)0, c,
                                                 false, false);
}

// One low-rank block over TWO 16-row tiles:
//   h <- tanh( (h @ Wc[256x32]) @ Wr^T[32x256] )
// Two independent WMMA streams per wave; B fragments shared across tiles.
// If LAST, the end-layer dot product is accumulated straight from D registers.
template<bool LAST>
__device__ __forceinline__ void lr_block2(bf16* hT0, bf16* hT1,
                                          bf16* uT0, bf16* uT1,
                                          const bf16* wcT,   // [32][WCPITCH] = (col*alpha)^T
                                          const bf16* wrk,   // [256][WRPITCH] = row
                                          int lane,
                                          const float* __restrict__ end_w,
                                          float* acc0Out, float* acc1Out) {
  const int hi  = lane >> 4;     // lane group (ISA A/B fragment K-split)
  const int n16 = lane & 15;

  // ---- matmul 1: U[16x32] = h[16x256] @ Wc[256x32], both tiles: 32 WMMAs ----
  v8f a00 = {}, a01 = {}, a10 = {}, a11 = {};
  const char* hrow0 = (const char*)(hT0 + n16 * HPITCH);
  const char* hrow1 = (const char*)(hT1 + n16 * HPITCH);
  const char* wc0   = (const char*)(wcT +  n16        * WCPITCH);
  const char* wc1   = (const char*)(wcT + (n16 + 16)  * WCPITCH);
#pragma unroll
  for (int kc = 0; kc < 8; ++kc) {
    const int o = kc * 64;
    // A layout: lanes<16 hold K {0..7,16..23}, lanes>=16 hold K {8..15,24..31}
    v16bf A0 = ld_frag(hrow0 + o + hi*16, hrow0 + o + 32 + hi*16);
    v16bf A1 = ld_frag(hrow1 + o + hi*16, hrow1 + o + 32 + hi*16);
    // B layout: same K split as A (lanes<16 K {0..7,16..23}, lanes>=16 K {8..15,24..31}); shared by both tiles
    v16bf B0 = ld_frag(wc0 + o + hi*16, wc0 + o + 32 + hi*16);
    v16bf B1 = ld_frag(wc1 + o + hi*16, wc1 + o + 32 + hi*16);
    a00 = wmma_bf16(A0, B0, a00);
    a01 = wmma_bf16(A0, B1, a01);
    a10 = wmma_bf16(A1, B0, a10);
    a11 = wmma_bf16(A1, B1, a11);
  }

  // Stage U (bf16, row-major [16][UPITCH]) so it can be re-read in A layout.
  // D layout: reg r, lane -> (m = r + 8*hi, n = lane%16)
#pragma unroll
  for (int r = 0; r < 8; ++r) {
    const int m = r + hi * 8;
    uT0[m * UPITCH + n16]      = (bf16)a00[r];
    uT0[m * UPITCH + n16 + 16] = (bf16)a01[r];
    uT1[m * UPITCH + n16]      = (bf16)a10[r];
    uT1[m * UPITCH + n16 + 16] = (bf16)a11[r];
  }

  // ---- matmul 2: V[16x256] = U[16x32] @ row^T[32x256], both tiles: 32 WMMAs ----
  const char* ur0 = (const char*)(uT0 + n16 * UPITCH);
  const char* ur1 = (const char*)(uT1 + n16 * UPITCH);
  v16bf UA0 = ld_frag(ur0 + hi*16, ur0 + 32 + hi*16);   // same-wave LDS RAW: in order
  v16bf UA1 = ld_frag(ur1 + hi*16, ur1 + 32 + hi*16);
  const char* wr_lane = (const char*)(wrk + n16 * WRPITCH);
#pragma unroll
  for (int nt = 0; nt < 16; ++nt) {
    const char* bp = wr_lane + nt * 16 * WRPITCH * 2;   // B[k][n] = row[16nt+n][k]
    v16bf B = ld_frag(bp + hi*16, bp + 32 + hi*16);     // shared by both tiles
    v8f z0 = {}, z1 = {};
    v8f v0 = wmma_bf16(UA0, B, z0);
    v8f v1 = wmma_bf16(UA1, B, z1);
    if (LAST) {
      const float ew = end_w[nt*16 + n16];
#pragma unroll
      for (int r = 0; r < 8; ++r) {
        acc0Out[r] += fast_tanh(v0[r]) * ew;
        acc1Out[r] += fast_tanh(v1[r]) * ew;
      }
    } else {
#pragma unroll
      for (int r = 0; r < 8; ++r) {
        hT0[(r + hi*8) * HPITCH + nt*16 + n16] = (bf16)fast_tanh(v0[r]);
        hT1[(r + hi*8) * HPITCH + nt*16 + n16] = (bf16)fast_tanh(v1[r]);
      }
    }
  }
}

__global__ void __launch_bounds__(WAVES * 32, 1) lrpinn_wmma_kernel(
    const float* __restrict__ x,  const float* __restrict__ t,
    const float* __restrict__ start_w, const float* __restrict__ start_b,
    const float* __restrict__ end_w,   const float* __restrict__ end_b,
    const float* __restrict__ col0, const float* __restrict__ col1, const float* __restrict__ col2,
    const float* __restrict__ row0, const float* __restrict__ row1, const float* __restrict__ row2,
    const float* __restrict__ a0,   const float* __restrict__ a1,   const float* __restrict__ a2,
    float* __restrict__ out, int nPairs)
{
  extern __shared__ __align__(16) char smem[];
  bf16* wcT    = (bf16*)(smem + OFF_WC);
  bf16* wrAll  = (bf16*)(smem + OFF_WR);
  const int lane   = threadIdx.x & 31;
  const int waveId = threadIdx.x >> 5;
  bf16* hT0 = (bf16*)(smem + OFF_H) + waveId * 32 * HPITCH;
  bf16* hT1 = hT0 + 16 * HPITCH;
  bf16* uT0 = (bf16*)(smem + OFF_U) + waveId * 32 * UPITCH;
  bf16* uT1 = uT0 + 16 * UPITCH;

  // ---- stage weights in LDS (bf16), once per workgroup ----
  const float* cols[3] = {col0, col1, col2};
  const float* rows[3] = {row0, row1, row2};
  const float* alps[3] = {a0, a1, a2};
  for (int k = 0; k < 3; ++k) {
    bf16* wck = wcT + k * 32 * WCPITCH;          // (col * alpha)^T : [32][256]
    for (int idx = threadIdx.x; idx < 32 * 256; idx += blockDim.x) {
      const int r = idx >> 8, hc = idx & 255;
      wck[r * WCPITCH + hc] = (bf16)(cols[k][hc * 32 + r] * alps[k][r]);
    }
    bf16* wrk = wrAll + k * 256 * WRPITCH;       // row : [256][32]
    for (int idx = threadIdx.x; idx < 256 * 32; idx += blockDim.x) {
      const int hr = idx >> 5, r = idx & 31;
      wrk[hr * WRPITCH + r] = (bf16)(rows[k][idx]);
    }
  }
  __syncthreads();

  // ---- two 16-row tiles (32 rows) per wave per iteration ----
  for (int pair = blockIdx.x * WAVES + waveId; pair < nPairs;
       pair += gridDim.x * WAVES) {
    const int rowBase = pair << 5;

    // start layer: h[m][j] = tanh(x*w0 + t*w1 + b). Each lane produces two
    // ADJACENT columns (packed ds_store_b32), for both tiles.
    const float xr0 = x[rowBase +      (lane & 15)];
    const float tr0 = t[rowBase +      (lane & 15)];
    const float xr1 = x[rowBase + 16 + (lane & 15)];
    const float tr1 = t[rowBase + 16 + (lane & 15)];
#pragma unroll
    for (int jc = 0; jc < 4; ++jc) {
      const int j = (jc << 6) + (lane << 1);                   // even column
      const float4 w  = *(const float4*)(start_w + 2 * j);     // w0_j,w1_j,w0_j+1,w1_j+1
      const float2 bb = *(const float2*)(start_b + j);
#pragma unroll
      for (int m = 0; m < 16; ++m) {
        const float xm0 = __shfl(xr0, m, 32), tm0 = __shfl(tr0, m, 32);
        const float xm1 = __shfl(xr1, m, 32), tm1 = __shfl(tr1, m, 32);
        PK2 c0, c1;
        c0.h[0] = (bf16)fast_tanh(fmaf(xm0, w.x, fmaf(tm0, w.y, bb.x)));
        c0.h[1] = (bf16)fast_tanh(fmaf(xm0, w.z, fmaf(tm0, w.w, bb.y)));
        c1.h[0] = (bf16)fast_tanh(fmaf(xm1, w.x, fmaf(tm1, w.y, bb.x)));
        c1.h[1] = (bf16)fast_tanh(fmaf(xm1, w.z, fmaf(tm1, w.w, bb.y)));
        *(unsigned int*)((char*)hT0 + (m * HPITCH + j) * 2) = c0.u;
        *(unsigned int*)((char*)hT1 + (m * HPITCH + j) * 2) = c1.u;
      }
    }

    float acc0[8] = {0.f,0.f,0.f,0.f,0.f,0.f,0.f,0.f};
    float acc1[8] = {0.f,0.f,0.f,0.f,0.f,0.f,0.f,0.f};
    lr_block2<false>(hT0, hT1, uT0, uT1, wcT + 0*32*WCPITCH, wrAll + 0*256*WRPITCH,
                     lane, end_w, acc0, acc1);
    lr_block2<false>(hT0, hT1, uT0, uT1, wcT + 1*32*WCPITCH, wrAll + 1*256*WRPITCH,
                     lane, end_w, acc0, acc1);
    lr_block2<true >(hT0, hT1, uT0, uT1, wcT + 2*32*WCPITCH, wrAll + 2*256*WRPITCH,
                     lane, end_w, acc0, acc1);

    // end layer: reduce the fused dot across each 16-lane group (n dimension)
    const float eb = end_b[0];
#pragma unroll
    for (int r = 0; r < 8; ++r) {
#pragma unroll
      for (int off = 8; off >= 1; off >>= 1) {
        acc0[r] += __shfl_xor(acc0[r], off, 32);
        acc1[r] += __shfl_xor(acc1[r], off, 32);
      }
    }
    if ((lane & 15) == 0) {
      const int mb = (lane >> 4) * 8;   // lane 0 -> rows 0..7, lane 16 -> rows 8..15
#pragma unroll
      for (int r = 0; r < 8; ++r) {
        out[rowBase +      mb + r] = acc0[r] + eb;
        out[rowBase + 16 + mb + r] = acc1[r] + eb;
      }
    }
  }
}

extern "C" void kernel_launch(void* const* d_in, const int* in_sizes, int n_in,
                              void* d_out, int out_size, void* d_ws, size_t ws_size,
                              hipStream_t stream) {
  const float* x       = (const float*)d_in[0];
  const float* t       = (const float*)d_in[1];
  const float* start_w = (const float*)d_in[2];
  const float* start_b = (const float*)d_in[3];
  const float* end_w   = (const float*)d_in[4];
  const float* end_b   = (const float*)d_in[5];
  const float* col0    = (const float*)d_in[6];
  const float* col1    = (const float*)d_in[7];
  const float* col2    = (const float*)d_in[8];
  const float* row0    = (const float*)d_in[9];
  const float* row1    = (const float*)d_in[10];
  const float* row2    = (const float*)d_in[11];
  const float* a0      = (const float*)d_in[12];
  const float* a1      = (const float*)d_in[13];
  const float* a2      = (const float*)d_in[14];

  const int N      = in_sizes[0];   // 262144
  const int nPairs = N >> 5;        // 8192 pairs of 16-row tiles

  dim3 grid(1024);
  dim3 block(WAVES * 32);           // 4 wave32 per workgroup
  lrpinn_wmma_kernel<<<grid, block, SMEM_BYTES, stream>>>(
      x, t, start_w, start_b, end_w, end_b,
      col0, col1, col2, row0, row1, row2, a0, a1, a2,
      (float*)d_out, nPairs);
}
